// VariationEncoder_46007689675484
// MI455X (gfx1250) — matrix operand order repaired and checked by reading, not compiled, not run on hardware
//
#include <hip/hip_runtime.h>
#include <hip/hip_bf16.h>

#define B_  16
#define T_  2048
#define D_  512
#define U_  256
#define N3_ 768          // 3*U
#define KTU_ 8           // U/32 K-tiles for hidden GEMMs
#define KTD_ 16          // D/32 K-tiles for input GEMM

typedef __attribute__((ext_vector_type(16))) __bf16       v16bf;
typedef __attribute__((ext_vector_type(8)))  float        v8f;
typedef __attribute__((ext_vector_type(8)))  unsigned int v8u;
typedef __attribute__((ext_vector_type(4)))  unsigned int v4u;

union FragA { v4u u4[2]; v16bf v; };
union FragB { v8u  u8;   v16bf v; };

static __device__ __forceinline__ unsigned short f2bf(float f) {
  unsigned int u = __float_as_uint(f);
  unsigned int r = u + 0x7fffu + ((u >> 16) & 1u);   // round-to-nearest-even
  return (unsigned short)(r >> 16);
}
static __device__ __forceinline__ float sigmoidf_(float x) {
  return 1.0f / (1.0f + __expf(-x));
}

// ---------------------------------------------------------------------------
// 1) f32 -> bf16 elementwise (for x)
// ---------------------------------------------------------------------------
__global__ void k_f32_to_bf16(const float* __restrict__ src,
                              unsigned short* __restrict__ dst, int n) {
  int i = blockIdx.x * blockDim.x + threadIdx.x;
  int stride = gridDim.x * blockDim.x;
  for (; i < n; i += stride) dst[i] = f2bf(src[i]);
}

// ---------------------------------------------------------------------------
// 2) Swizzle a KxN f32 weight matrix into WMMA B-fragment order (bf16).
//    Fragment (nt, kt): 32 lanes x 8 dwords, contiguous per lane.
//    Lane L: column N = nt*16 + (L&15); dword j packs K = kbase+2j+8*(j>=4), +1
//    with kbase = kt*32 + 8*(L>=16).   (ISA 7.12.2, 16-bit B 32x16)
// ---------------------------------------------------------------------------
__global__ void k_swizzle_w(const float* __restrict__ w,
                            unsigned int* __restrict__ out, int K, int N) {
  int KT = K >> 5;
  int total = (K >> 5) * (N >> 4) * 32;
  int tid = blockIdx.x * blockDim.x + threadIdx.x;
  if (tid >= total) return;
  int lane = tid & 31;
  int frag = tid >> 5;            // frag = nt*KT + kt
  int kt = frag % KT;
  int nt = frag / KT;
  int n = nt * 16 + (lane & 15);
  int kbase = kt * 32 + ((lane & 16) ? 8 : 0);
  unsigned int* o = out + (size_t)(frag * 32 + lane) * 8;
#pragma unroll
  for (int j = 0; j < 8; ++j) {
    int k0 = kbase + 2 * j + ((j >= 4) ? 8 : 0);
    unsigned int lo = f2bf(w[(size_t)k0 * N + n]);
    unsigned int hi = f2bf(w[(size_t)(k0 + 1) * N + n]);
    o[j] = lo | (hi << 16);
  }
}

// ---------------------------------------------------------------------------
// 3) XP = x @ W0 + b0   for one direction.  (B*T, 512) @ (512, 768) -> f32
//    grid = (2048 Mtiles, 12 Ntile-groups), block = 128 (4 waves, 1 Ntile/wave)
// ---------------------------------------------------------------------------
__global__ void __launch_bounds__(128)
k_xp_gemm(const unsigned short* __restrict__ xa,   // (B*T, 512) bf16 row-major
          const unsigned int*  __restrict__ w0s,   // swizzled 512x768
          const float*         __restrict__ bias,  // 768
          float*               __restrict__ xp) {  // (B*T, 768) f32
  const int lane  = threadIdx.x & 31;
  const int wave  = threadIdx.x >> 5;
  const int mtile = blockIdx.x;
  const int ntile = blockIdx.y * 4 + wave;
  const int mrow  = lane & 15;
  const int koff  = (lane & 16) ? 8 : 0;
  const int ncol  = ntile * 16 + mrow;

  const float bv = bias[ncol];
  v8f acc;
#pragma unroll
  for (int r = 0; r < 8; ++r) acc[r] = bv;

  const unsigned short* arow = xa + (size_t)(mtile * 16 + mrow) * D_;
#pragma unroll 4
  for (int kt = 0; kt < KTD_; ++kt) {
    FragA a;
    const unsigned short* ap = arow + kt * 32 + koff;
    a.u4[0] = *(const v4u*)(ap);
    a.u4[1] = *(const v4u*)(ap + 16);
    FragB b;
    b.u8 = *(const v8u*)(w0s + (size_t)((ntile * KTD_ + kt) * 32 + lane) * 8);
    acc = __builtin_amdgcn_wmma_f32_16x16x32_bf16(false, a.v, false, b.v,
                                                  (short)0, acc, false, false);
  }
  const int mrow0 = mtile * 16 + ((lane & 16) ? 8 : 0);
#pragma unroll
  for (int r = 0; r < 8; ++r)
    xp[(size_t)(mrow0 + r) * N3_ + ncol] = acc[r];
}

// ---------------------------------------------------------------------------
// 4) Persistent GRU scan: 2 blocks (dir 0 = fwd, dir 1 = bwd), 512 threads.
//    LDS: h0/h1 state (f32 + bf16 mirrors) and 16x768 f32 GEMM staging.
// ---------------------------------------------------------------------------
__global__ void __launch_bounds__(512)
k_gru_scan(const float* __restrict__ xp_f, const float* __restrict__ xp_b,
           const unsigned int* __restrict__ u0s_f,
           const unsigned int* __restrict__ w1s_f,
           const unsigned int* __restrict__ u1s_f,
           const unsigned int* __restrict__ u0s_b,
           const unsigned int* __restrict__ w1s_b,
           const unsigned int* __restrict__ u1s_b,
           const float* __restrict__ b1f_, const float* __restrict__ b1b_,
           const unsigned char* __restrict__ mask,
           float* __restrict__ out) {
  extern __shared__ char smem[];
  unsigned short* h0bs = (unsigned short*)(smem);            //  8 KB
  unsigned short* h1bs = (unsigned short*)(smem + 8192);     //  8 KB
  float* h0f = (float*)(smem + 16384);                       // 16 KB
  float* h1f = (float*)(smem + 32768);                       // 16 KB
  float* HU0 = (float*)(smem + 49152);                       // 48 KB
  float* XP1 = (float*)(smem + 98304);                       // 48 KB
  float* HU1 = (float*)(smem + 147456);                      // 48 KB (total 192 KB)

  const int dir = blockIdx.x;
  const float*        xp  = dir ? xp_b  : xp_f;
  const unsigned int* u0s = dir ? u0s_b : u0s_f;
  const unsigned int* w1s = dir ? w1s_b : w1s_f;
  const unsigned int* u1s = dir ? u1s_b : u1s_f;
  const float*        b1  = dir ? b1b_  : b1f_;

  const int lane = threadIdx.x & 31;
  const int wave = threadIdx.x >> 5;      // 0..15 -> 3 N-tiles each (48 total)
  const int mrow = lane & 15;
  const int koff = (lane & 16) ? 8 : 0;   // A-frag K offset (halves)
  const int mhi  = (lane & 16) ? 8 : 0;   // C-frag row offset

  // elementwise-phase indexing: 4096 elems, 8 per thread, fixed batch row
  const int ebase = threadIdx.x * 8;
  const int eb    = ebase >> 8;           // batch 0..15
  const int eu0   = ebase & 255;          // unit base

  for (int i = threadIdx.x; i < B_ * U_; i += 512) {
    h0f[i] = 0.0f; h1f[i] = 0.0f; h0bs[i] = 0; h1bs[i] = 0;
  }
  __syncthreads();

  for (int t = 0; t < T_; ++t) {
    const int tsrc = dir ? (T_ - 1 - t) : t;

    // ---- Phase A: HU0 = h0 @ U0 ------------------------------------------
#pragma unroll
    for (int nn = 0; nn < 3; ++nn) {
      const int ntile = wave * 3 + nn;
      const int ncol  = ntile * 16 + mrow;
      v8f acc = {};
#pragma unroll
      for (int kt = 0; kt < KTU_; ++kt) {
        FragA a;
        const unsigned short* ap = h0bs + mrow * U_ + kt * 32 + koff;
        a.u4[0] = *(const v4u*)(ap);
        a.u4[1] = *(const v4u*)(ap + 16);
        FragB b;
        b.u8 = *(const v8u*)(u0s + (size_t)((ntile * KTU_ + kt) * 32 + lane) * 8);
        acc = __builtin_amdgcn_wmma_f32_16x16x32_bf16(false, a.v, false, b.v,
                                                      (short)0, acc, false, false);
      }
#pragma unroll
      for (int r = 0; r < 8; ++r) HU0[(mhi + r) * N3_ + ncol] = acc[r];
    }
    __syncthreads();

    // ---- Phase B: layer-0 gates ------------------------------------------
    {
      const size_t row = (size_t)eb * T_ + tsrc;
      const float* xrow = xp + row * N3_;
      const float* hrow = HU0 + eb * N3_;
      const float mt = mask[row] ? 1.0f : 0.0f;
      if (t + 1 < T_) {  // prefetch next step's xp row into cache
        const size_t nrow = (size_t)eb * T_ + (dir ? (T_ - 2 - t) : (t + 1));
        __builtin_prefetch(xp + nrow * N3_ + eu0, 0, 1);
      }
#pragma unroll
      for (int e = 0; e < 8; ++e) {
        const int u = eu0 + e;
        const float z  = sigmoidf_(xrow[u]        + hrow[u]);
        const float r  = sigmoidf_(xrow[U_ + u]   + hrow[U_ + u]);
        const float hh = tanhf(xrow[2 * U_ + u] + r * hrow[2 * U_ + u]);
        const int   hi = eb * U_ + u;
        const float hold = h0f[hi];
        float hn = z * hold + (1.0f - z) * hh;
        hn = mt * hn + (1.0f - mt) * hold;
        h0f[hi]  = hn;
        h0bs[hi] = f2bf(hn);
      }
    }
    __syncthreads();

    // ---- Phase C: XP1 = h0_new @ W1 + b1 ; HU1 = h1 @ U1 -----------------
#pragma unroll
    for (int nn = 0; nn < 3; ++nn) {
      const int ntile = wave * 3 + nn;
      const int ncol  = ntile * 16 + mrow;
      const float bv = b1[ncol];
      v8f accX;
#pragma unroll
      for (int r = 0; r < 8; ++r) accX[r] = bv;
      v8f accH = {};
#pragma unroll
      for (int kt = 0; kt < KTU_; ++kt) {
        const int aoff = mrow * U_ + kt * 32 + koff;
        FragA a0, a1;
        a0.u4[0] = *(const v4u*)(h0bs + aoff);
        a0.u4[1] = *(const v4u*)(h0bs + aoff + 16);
        a1.u4[0] = *(const v4u*)(h1bs + aoff);
        a1.u4[1] = *(const v4u*)(h1bs + aoff + 16);
        FragB bw, bu;
        const size_t fbase = (size_t)((ntile * KTU_ + kt) * 32 + lane) * 8;
        bw.u8 = *(const v8u*)(w1s + fbase);
        bu.u8 = *(const v8u*)(u1s + fbase);
        accX = __builtin_amdgcn_wmma_f32_16x16x32_bf16(false, a0.v, false, bw.v,
                                                       (short)0, accX, false, false);
        accH = __builtin_amdgcn_wmma_f32_16x16x32_bf16(false, a1.v, false, bu.v,
                                                       (short)0, accH, false, false);
      }
#pragma unroll
      for (int r = 0; r < 8; ++r) {
        XP1[(mhi + r) * N3_ + ncol] = accX[r];
        HU1[(mhi + r) * N3_ + ncol] = accH[r];
      }
    }
    __syncthreads();

    // ---- Phase D: layer-1 gates + output store ---------------------------
    {
      const size_t row = (size_t)eb * T_ + tsrc;
      const float* xrow = XP1 + eb * N3_;
      const float* hrow = HU1 + eb * N3_;
      const float mt = mask[row] ? 1.0f : 0.0f;
      float* orow = out + row * 512 + dir * 256;
#pragma unroll
      for (int e = 0; e < 8; ++e) {
        const int u = eu0 + e;
        const float z  = sigmoidf_(xrow[u]        + hrow[u]);
        const float r  = sigmoidf_(xrow[U_ + u]   + hrow[U_ + u]);
        const float hh = tanhf(xrow[2 * U_ + u] + r * hrow[2 * U_ + u]);
        const int   hi = eb * U_ + u;
        const float hold = h1f[hi];
        float hn = z * hold + (1.0f - z) * hh;
        hn = mt * hn + (1.0f - mt) * hold;
        h1f[hi]  = hn;
        h1bs[hi] = f2bf(hn);
        orow[u]  = hn;
      }
    }
    __syncthreads();
  }

  // ---- final states: st_f0, st_f1, st_b0, st_b1 --------------------------
  {
    float* st = out + (size_t)B_ * T_ * 512 + (size_t)dir * 2 * (B_ * U_);
#pragma unroll
    for (int e = 0; e < 8; ++e) {
      st[ebase + e]            = h0f[ebase + e];
      st[B_ * U_ + ebase + e]  = h1f[ebase + e];
    }
  }
}

// ---------------------------------------------------------------------------
extern "C" void kernel_launch(void* const* d_in, const int* in_sizes, int n_in,
                              void* d_out, int out_size, void* d_ws, size_t ws_size,
                              hipStream_t stream) {
  (void)in_sizes; (void)n_in; (void)out_size; (void)ws_size;
  const float*         x    = (const float*)d_in[0];
  const unsigned char* mask = (const unsigned char*)d_in[1];
  const float* W0f = (const float*)d_in[2];
  const float* U0f = (const float*)d_in[3];
  const float* b0f = (const float*)d_in[4];
  const float* W1f = (const float*)d_in[5];
  const float* U1f = (const float*)d_in[6];
  const float* b1f = (const float*)d_in[7];
  const float* W0b = (const float*)d_in[8];
  const float* U0b = (const float*)d_in[9];
  const float* b0b = (const float*)d_in[10];
  const float* W1b = (const float*)d_in[11];
  const float* U1b = (const float*)d_in[12];
  const float* b1b = (const float*)d_in[13];
  float* out = (float*)d_out;

  char* ws = (char*)d_ws;
  size_t off = 0;
  unsigned short* XA  = (unsigned short*)(ws + off); off += (size_t)B_ * T_ * D_ * 2;
  float* XPf = (float*)(ws + off); off += (size_t)B_ * T_ * N3_ * 4;
  float* XPb = (float*)(ws + off); off += (size_t)B_ * T_ * N3_ * 4;
  unsigned int* W0sf = (unsigned int*)(ws + off); off += (size_t)D_ * N3_ * 2;
  unsigned int* W0sb = (unsigned int*)(ws + off); off += (size_t)D_ * N3_ * 2;
  unsigned int* U0sf = (unsigned int*)(ws + off); off += (size_t)U_ * N3_ * 2;
  unsigned int* W1sf = (unsigned int*)(ws + off); off += (size_t)U_ * N3_ * 2;
  unsigned int* U1sf = (unsigned int*)(ws + off); off += (size_t)U_ * N3_ * 2;
  unsigned int* U0sb = (unsigned int*)(ws + off); off += (size_t)U_ * N3_ * 2;
  unsigned int* W1sb = (unsigned int*)(ws + off); off += (size_t)U_ * N3_ * 2;
  unsigned int* U1sb = (unsigned int*)(ws + off); off += (size_t)U_ * N3_ * 2;

  // 1) x -> bf16
  const int nx = B_ * T_ * D_;
  k_f32_to_bf16<<<4096, 256, 0, stream>>>(x, XA, nx);

  // 2) swizzle all weights into WMMA B-fragment order
  const int tw512 = (D_ / 32) * (N3_ / 16) * 32;   // 24576
  const int tw256 = (U_ / 32) * (N3_ / 16) * 32;   // 12288
  k_swizzle_w<<<(tw512 + 255) / 256, 256, 0, stream>>>(W0f, W0sf, D_, N3_);
  k_swizzle_w<<<(tw512 + 255) / 256, 256, 0, stream>>>(W0b, W0sb, D_, N3_);
  k_swizzle_w<<<(tw256 + 255) / 256, 256, 0, stream>>>(U0f, U0sf, U_, N3_);
  k_swizzle_w<<<(tw256 + 255) / 256, 256, 0, stream>>>(W1f, W1sf, U_, N3_);
  k_swizzle_w<<<(tw256 + 255) / 256, 256, 0, stream>>>(U1f, U1sf, U_, N3_);
  k_swizzle_w<<<(tw256 + 255) / 256, 256, 0, stream>>>(U0b, U0sb, U_, N3_);
  k_swizzle_w<<<(tw256 + 255) / 256, 256, 0, stream>>>(W1b, W1sb, U_, N3_);
  k_swizzle_w<<<(tw256 + 255) / 256, 256, 0, stream>>>(U1b, U1sb, U_, N3_);

  // 3) XP = x @ W0 + b0 for both directions (parallel WMMA GEMM)
  dim3 ggrid((B_ * T_) / 16, N3_ / 64);            // 2048 x 12
  k_xp_gemm<<<ggrid, 128, 0, stream>>>(XA, W0sf, b0f, XPf);
  k_xp_gemm<<<ggrid, 128, 0, stream>>>(XA, W0sb, b0b, XPb);

  // 4) persistent bidirectional scan (1 WGP-resident workgroup per direction)
  k_gru_scan<<<dim3(2), dim3(512), 196608, stream>>>(
      XPf, XPb, U0sf, W1sf, U1sf, U0sb, W1sb, U1sb, b1f, b1b, mask, out);
}
